// Model_61959198212555
// MI455X (gfx1250) — compile-verified, matrix operand order checked by reading, not run on hardware
//
#include <hip/hip_runtime.h>

// ---------------------------------------------------------------------------
// Model dims
// ---------------------------------------------------------------------------
#define Bb 32
#define Cc 1024
#define Gg 16
#define Ee 16384
#define Tt 24
#define F_IN 8
#define X_EM 32
#define LOC_EM 12
#define DATE_EM 4
#define EDGE_H 12
#define GNN_H 32
#define PREDD 24
#define Pp 240                    // G*(G-1)
#define NNODE (Bb*Cc)             // 32768
#define NGRP  (Bb*Gg)             // 512
#define EGRP  (Bb*Pp)             // 7680
#define EGLOB (Bb*Ee)             // 524288

typedef _Float16 half16 __attribute__((ext_vector_type(16)));
typedef _Float16 half8  __attribute__((ext_vector_type(8)));
typedef float    float8 __attribute__((ext_vector_type(8)));

#define LOG2E 1.4426950408889634f

// Branch-free activations on raw HW transcendentals (v_exp_f32 / v_rcp_f32).
// Keeps EXEC all-1s through the recurrent loop (required around WMMA anyway).
__device__ __forceinline__ float fast_sigm(float v)
{
    v = fminf(fmaxf(v, -30.0f), 30.0f);
    const float e = __builtin_amdgcn_exp2f(-v * LOG2E);
    return __builtin_amdgcn_rcpf(1.0f + e);
}

__device__ __forceinline__ float fast_tanh(float v)
{
    v = fminf(fmaxf(v, -15.0f), 15.0f);
    const float e = __builtin_amdgcn_exp2f(-2.0f * LOG2E * v);  // exp(-2v)
    return (1.0f - e) * __builtin_amdgcn_rcpf(1.0f + e);
}

// ---------------------------------------------------------------------------
// LSTM over T=24 steps; one wave32 handles 16 sequences; gates via WMMA f16.
//   A (16x32 f16): lane -> M (sequence), regs -> K  (hidden / padded features)
//   B (32x16 f16): lane -> N (gate unit), regs -> K
//   C/D (16x16 f32): reg r -> M = r + 8*(lane>=16), N = lane&15
// K mapping within v16h element i:  K = i + 8*hi + (i>=8 ? 8 : 0)
// ---------------------------------------------------------------------------
__launch_bounds__(128)
__global__ void lstm_kernel(const float* __restrict__ x,
                            const float* __restrict__ Wih,
                            const float* __restrict__ Whh,
                            const float* __restrict__ bih,
                            const float* __restrict__ bhh,
                            float* __restrict__ xh)
{
    __shared__ __align__(16) _Float16 lds_h[4][16 * 32];

    const int lane = threadIdx.x & 31;
    const int wave = threadIdx.x >> 5;
    const int hi   = lane >> 4;
    const int lm   = lane & 15;
    const int seq0 = (blockIdx.x * 4 + wave) * 16;

    // Load B operands (weights) once, f32 -> f16
    half16 b_h[8], b_x[8];
    float  bias[8];
#pragma unroll
    for (int nt = 0; nt < 8; ++nt) {
        const int n = nt * 16 + lm;            // gate unit (0..127)
        half16 bh, bx;
#pragma unroll
        for (int i = 0; i < 16; ++i) {
            const int k = i + 8 * hi + ((i >= 8) ? 8 : 0);
            bh[i] = (_Float16)Whh[n * 32 + k];
            bx[i] = (k < F_IN) ? (_Float16)Wih[n * F_IN + k] : (_Float16)0.0f;
        }
        b_h[nt] = bh;
        b_x[nt] = bx;
        bias[nt] = bih[n] + bhh[n];
    }

    half16 a_h;
#pragma unroll
    for (int i = 0; i < 16; ++i) a_h[i] = (_Float16)0.0f;

    float cst[2][8];
#pragma unroll
    for (int jt = 0; jt < 2; ++jt)
#pragma unroll
        for (int r = 0; r < 8; ++r) cst[jt][r] = 0.0f;

    _Float16* myl = lds_h[wave];

    for (int t = 0; t < Tt; ++t) {
        // A_x: only lanes 0..15 carry K=0..7 (features); everything else zero
        half16 a_x;
        if (hi == 0) {
            const float* xp = x + ((long)(seq0 + lm) * Tt + t) * F_IN;
#pragma unroll
            for (int i = 0; i < 8; ++i)  a_x[i] = (_Float16)xp[i];
#pragma unroll
            for (int i = 8; i < 16; ++i) a_x[i] = (_Float16)0.0f;
        } else {
#pragma unroll
            for (int i = 0; i < 16; ++i) a_x[i] = (_Float16)0.0f;
        }

#pragma unroll
        for (int jt = 0; jt < 2; ++jt) {
            float8 ai, af, ag, ao;
#pragma unroll
            for (int r = 0; r < 8; ++r) {
                ai[r] = bias[jt];
                af[r] = bias[2 + jt];
                ag[r] = bias[4 + jt];
                ao[r] = bias[6 + jt];
            }
            ai = __builtin_amdgcn_wmma_f32_16x16x32_f16(false, a_x, false, b_x[jt],     (short)0, ai, false, false);
            ai = __builtin_amdgcn_wmma_f32_16x16x32_f16(false, a_h, false, b_h[jt],     (short)0, ai, false, false);
            af = __builtin_amdgcn_wmma_f32_16x16x32_f16(false, a_x, false, b_x[2 + jt], (short)0, af, false, false);
            af = __builtin_amdgcn_wmma_f32_16x16x32_f16(false, a_h, false, b_h[2 + jt], (short)0, af, false, false);
            ag = __builtin_amdgcn_wmma_f32_16x16x32_f16(false, a_x, false, b_x[4 + jt], (short)0, ag, false, false);
            ag = __builtin_amdgcn_wmma_f32_16x16x32_f16(false, a_h, false, b_h[4 + jt], (short)0, ag, false, false);
            ao = __builtin_amdgcn_wmma_f32_16x16x32_f16(false, a_x, false, b_x[6 + jt], (short)0, ao, false, false);
            ao = __builtin_amdgcn_wmma_f32_16x16x32_f16(false, a_h, false, b_h[6 + jt], (short)0, ao, false, false);

#pragma unroll
            for (int r = 0; r < 8; ++r) {
                const float cv = fast_sigm(af[r]) * cst[jt][r] + fast_sigm(ai[r]) * fast_tanh(ag[r]);
                cst[jt][r] = cv;
                const float hv = fast_sigm(ao[r]) * fast_tanh(cv);
                // D layout -> LDS [seq_local][hidden]
                myl[(r + 8 * hi) * 32 + jt * 16 + lm] = (_Float16)hv;
            }
        }

        // Reload h in A layout: lane -> seq = lm; regs -> K per WMMA striping
        const half8 lo8 = *(const half8*)&myl[lm * 32 + 8 * hi];
        const half8 hi8 = *(const half8*)&myl[lm * 32 + 16 + 8 * hi];
#pragma unroll
        for (int i = 0; i < 8; ++i) { a_h[i] = lo8[i]; a_h[8 + i] = hi8[i]; }
    }

    // Emit final hidden state: lane writes seq=lm, units hi*16 .. hi*16+15
#pragma unroll
    for (int i = 0; i < 16; ++i)
        xh[(long)(seq0 + lm) * 32 + hi * 16 + i] = (float)myl[lm * 32 + hi * 16 + i];
}

// ---------------------------------------------------------------------------
// Small pipeline kernels (memory-bound stages, plain f32 VALU)
// ---------------------------------------------------------------------------
__global__ void softmax_kernel(const float* __restrict__ w, float* __restrict__ wsm)
{
    int c = blockIdx.x * blockDim.x + threadIdx.x;
    if (c >= Cc) return;
    float m = -1e30f;
    for (int g = 0; g < Gg; ++g) m = fmaxf(m, w[c * Gg + g]);
    float e[Gg], s = 0.0f;
    for (int g = 0; g < Gg; ++g) {
        e[g] = __builtin_amdgcn_exp2f((w[c * Gg + g] - m) * LOG2E);
        s += e[g];
    }
    const float inv = 1.0f / s;
    for (int g = 0; g < Gg; ++g) wsm[c * Gg + g] = e[g] * inv;
}

__global__ void xloc_kernel(const float* __restrict__ xh, const float* __restrict__ loc,
                            const float* __restrict__ locW, const float* __restrict__ locb,
                            float* __restrict__ x_loc)
{
    int n = blockIdx.x * blockDim.x + threadIdx.x;
    if (n >= NNODE) return;
    const float l0 = loc[n * 2 + 0], l1 = loc[n * 2 + 1];
    float* o = x_loc + (long)n * 44;
    for (int k = 0; k < 32; ++k) o[k] = xh[(long)n * 32 + k];
    for (int k = 0; k < 12; ++k) o[32 + k] = locb[k] + locW[k * 2] * l0 + locW[k * 2 + 1] * l1;
}

// dst[b][g][d] = sum_c wcg[c][g] * src[b][c][d]
__global__ void einsum_cg_kernel(const float* __restrict__ wcg, const float* __restrict__ src,
                                 float* __restrict__ dst, int D)
{
    const int bg = blockIdx.x;
    const int b = bg / Gg, g = bg % Gg;
    const int d = threadIdx.x;
    if (d >= D) return;
    float acc = 0.0f;
    const float* sp = src + (long)b * Cc * D + d;
    for (int c = 0; c < Cc; ++c) acc += wcg[c * Gg + g] * sp[(long)c * D];
    dst[(long)bg * D + d] = acc;
}

__global__ void uem_kernel(const int* __restrict__ u, const float* __restrict__ e1,
                           const float* __restrict__ e2, const float* __restrict__ e3,
                           float* __restrict__ uem)
{
    int b = blockIdx.x * blockDim.x + threadIdx.x;
    if (b >= Bb) return;
    const int a0 = u[b * 3 + 0], a1 = u[b * 3 + 1], a2 = u[b * 3 + 2];
    for (int k = 0; k < 4; ++k) {
        uem[b * 12 + k]     = e1[a0 * 4 + k];
        uem[b * 12 + 4 + k] = e2[a1 * 4 + k];
        uem[b * 12 + 8 + k] = e3[a2 * 4 + k];
    }
}

__global__ void edgeinf_kernel(const float* __restrict__ g_x, const float* __restrict__ uem,
                               const float* __restrict__ W, const float* __restrict__ bb,
                               float* __restrict__ g_ew)
{
    int idx = blockIdx.x * blockDim.x + threadIdx.x;
    if (idx >= Bb * Pp * EDGE_H) return;
    const int o = idx % EDGE_H;
    const int p = (idx / EDGE_H) % Pp;
    const int b = idx / (EDGE_H * Pp);
    const int i = p / 15, q = p % 15;
    const int j = (q < i) ? q : q + 1;
    const float* wi = W + o * 100;
    const float* gi = g_x + (long)(b * Gg + i) * 44;
    const float* gj = g_x + (long)(b * Gg + j) * 44;
    float acc = bb[o];
    for (int k = 0; k < 44; ++k) acc += wi[k] * gi[k];
    for (int k = 0; k < 44; ++k) acc += wi[44 + k] * gj[k];
    for (int k = 0; k < 12; ++k) acc += wi[88 + k] * uem[b * 12 + k];
    g_ew[(long)(b * Pp + p) * EDGE_H + o] = fmaxf(acc, 0.0f);
}

__global__ void group_edges_kernel(int* __restrict__ grow, int* __restrict__ gcol)
{
    int idx = blockIdx.x * blockDim.x + threadIdx.x;
    if (idx >= EGRP) return;
    const int b = idx / Pp, p = idx % Pp;
    const int i = p / 15, q = p % 15;
    const int j = (q < i) ? q : q + 1;
    grow[idx] = b * Gg + i;
    gcol[idx] = b * Gg + j;
}

__global__ void global_edges_kernel(const int* __restrict__ ei, int* __restrict__ row,
                                    int* __restrict__ col)
{
    int idx = blockIdx.x * blockDim.x + threadIdx.x;
    if (idx >= EGLOB) return;
    const int b = idx / Ee, e = idx % Ee;
    row[idx] = b * Cc + ei[(long)(b * 2 + 0) * Ee + e];
    col[idx] = b * Cc + ei[(long)(b * 2 + 1) * Ee + e];
}

__global__ void zero_kernel(float* __restrict__ p, long n)
{
    long i = (long)blockIdx.x * blockDim.x + threadIdx.x;
    if (i < n) p[i] = 0.0f;
}

// edge MLP + scatter-add (fused): h = relu(W1 [x[row], eattr] + b1); s[col] += h
__global__ void edge_scatter_kernel(const float* __restrict__ xn, int xdim,
                                    const float* __restrict__ ea, int edim,
                                    const int* __restrict__ row, const int* __restrict__ col,
                                    const float* __restrict__ W1, const float* __restrict__ b1,
                                    float* __restrict__ s, float* __restrict__ cnt, long nE)
{
    long idx = (long)blockIdx.x * blockDim.x + threadIdx.x;
    if (idx >= nE * GNN_H) return;
    const int o = (int)(idx & (GNN_H - 1));
    const long e = idx >> 5;
    const int K = xdim + edim;
    const float* wr = W1 + o * K;
    const float* xp = xn + (long)row[e] * xdim;
    float acc = b1[o];
    for (int k = 0; k < xdim; ++k) acc += wr[k] * xp[k];
    const float* ep = ea + e * edim;
    for (int k = 0; k < edim; ++k) acc += wr[xdim + k] * ep[k];
    const float v = fmaxf(acc, 0.0f);
    atomicAdd(&s[(long)col[e] * GNN_H + o], v);
    if (o == 0) atomicAdd(&cnt[col[e]], 1.0f);
}

// out = relu(W2 [x, s/max(cnt,1)] + b2)
__global__ void node_mlp_kernel(const float* __restrict__ xn, int xdim,
                                const float* __restrict__ s, const float* __restrict__ cnt,
                                const float* __restrict__ W2, const float* __restrict__ b2,
                                float* __restrict__ out, long N)
{
    long idx = (long)blockIdx.x * blockDim.x + threadIdx.x;
    if (idx >= N * GNN_H) return;
    const int o = (int)(idx & (GNN_H - 1));
    const long n = idx >> 5;
    const int K = xdim + GNN_H;
    const float* wr = W2 + o * K;
    const float* xp = xn + n * xdim;
    float acc = b2[o];
    for (int k = 0; k < xdim; ++k) acc += wr[k] * xp[k];
    const float inv = 1.0f / fmaxf(cnt[n], 1.0f);
    const float* sp = s + n * GNN_H;
    for (int k = 0; k < GNN_H; ++k) acc += wr[xdim + k] * (sp[k] * inv);
    out[n * GNN_H + o] = fmaxf(acc, 0.0f);
}

// out[b,c] = concat(xfirst[b,c] (32), sum_g wcg[c][g]*gv[b,g,:] (32))
__global__ void combine_kernel(const float* __restrict__ xfirst, const float* __restrict__ wcg,
                               const float* __restrict__ gv, float* __restrict__ out)
{
    int n = blockIdx.x * blockDim.x + threadIdx.x;
    if (n >= NNODE) return;
    const int b = n / Cc, c = n % Cc;
    float* op = out + (long)n * 64;
    for (int k = 0; k < 32; ++k) op[k] = xfirst[(long)n * 32 + k];
    const float* wv = wcg + c * Gg;
    for (int d = 0; d < 32; ++d) {
        float acc = 0.0f;
        for (int g = 0; g < Gg; ++g) acc += wv[g] * gv[(long)(b * Gg + g) * 32 + d];
        op[32 + d] = acc;
    }
}

__global__ void linear_kernel(const float* __restrict__ xn, const float* __restrict__ W,
                              const float* __restrict__ bb, float* __restrict__ out,
                              long N, int K, int O, int do_relu)
{
    long idx = (long)blockIdx.x * blockDim.x + threadIdx.x;
    if (idx >= N * O) return;
    const int o = (int)(idx % O);
    const long n = idx / O;
    const float* wr = W + o * K;
    const float* xp = xn + n * K;
    float acc = bb[o];
    for (int k = 0; k < K; ++k) acc += wr[k] * xp[k];
    out[n * O + o] = do_relu ? fmaxf(acc, 0.0f) : acc;
}

// ---------------------------------------------------------------------------
// Launcher
// ---------------------------------------------------------------------------
extern "C" void kernel_launch(void* const* d_in, const int* in_sizes, int n_in,
                              void* d_out, int out_size, void* d_ws, size_t ws_size,
                              hipStream_t stream)
{
    (void)in_sizes; (void)n_in; (void)out_size; (void)ws_size;

    const float* x      = (const float*)d_in[0];
    const int*   u      = (const int*)  d_in[1];
    const int*   eidx   = (const int*)  d_in[2];
    const float* edge_w = (const float*)d_in[3];
    const float* loc    = (const float*)d_in[4];
    const float* Wih    = (const float*)d_in[5];
    const float* Whh    = (const float*)d_in[6];
    const float* bih    = (const float*)d_in[7];
    const float* bhh    = (const float*)d_in[8];
    const float* wraw   = (const float*)d_in[9];
    const float* locW   = (const float*)d_in[10];
    const float* locb   = (const float*)d_in[11];
    const float* emb1   = (const float*)d_in[12];
    const float* emb2   = (const float*)d_in[13];
    const float* emb3   = (const float*)d_in[14];
    const float* einfW  = (const float*)d_in[15];
    const float* einfb  = (const float*)d_in[16];
    // group_gnn[0]: 17..20, group_gnn[1]: 21..24
    // global_gnn[0]: 25..28, global_gnn[1]: 29..32
    // dec_xemb: 33,34
    // dec_group_gnn: 35..42, dec_global_gnn: 43..50
    // pred: 51..54
    auto F = [&](int i) { return (const float*)d_in[i]; };

    float* wsf = (float*)d_ws;
    size_t off = 0;
    auto alloc = [&](size_t n) { float* p = wsf + off; off += n; return p; };

    float* xh    = alloc((size_t)NNODE * 32);
    float* wsm   = alloc((size_t)Cc * Gg);
    float* x_loc = alloc((size_t)NNODE * 44);
    float* g_x   = alloc((size_t)NGRP * 44);
    float* u_em  = alloc((size_t)Bb * 12);
    float* g_ew  = alloc((size_t)EGRP * EDGE_H);
    float* gx1   = alloc((size_t)NGRP * 32);
    float* gx2   = alloc((size_t)NGRP * 32);
    float* bufA  = alloc((size_t)NNODE * 64);   // new_x, later dn
    float* bufB  = alloc((size_t)NNODE * 32);   // nx1 / dnx1 / h1
    float* bufC  = alloc((size_t)NNODE * 32);   // nx2 / dnx2
    float* dx    = alloc((size_t)NNODE * 32);
    float* dg    = alloc((size_t)NGRP * 32);
    float* sbuf  = alloc((size_t)NNODE * 32);   // scatter accumulator (reused)
    float* cnt   = alloc((size_t)NNODE);
    int*   g_row = (int*)alloc((size_t)EGRP);
    int*   g_col = (int*)alloc((size_t)EGRP);
    int*   grow  = (int*)alloc((size_t)EGLOB);
    int*   gcol  = (int*)alloc((size_t)EGLOB);
    float* outp  = (float*)d_out;

    // 1) LSTM encoder (WMMA)
    lstm_kernel<<<NNODE / 64, 128, 0, stream>>>(x, Wih, Whh, bih, bhh, xh);

    // 2) softmax(w), loc embed + concat, g_x einsum, date embeds, edge inference
    softmax_kernel<<<(Cc + 255) / 256, 256, 0, stream>>>(wraw, wsm);
    xloc_kernel<<<(NNODE + 255) / 256, 256, 0, stream>>>(xh, loc, locW, locb, x_loc);
    einsum_cg_kernel<<<Bb * Gg, 64, 0, stream>>>(wsm, x_loc, g_x, 44);
    uem_kernel<<<1, 32, 0, stream>>>(u, emb1, emb2, emb3, u_em);
    edgeinf_kernel<<<(Bb * Pp * EDGE_H + 255) / 256, 256, 0, stream>>>(g_x, u_em, einfW, einfb, g_ew);

    // edge index arrays
    group_edges_kernel<<<(EGRP + 255) / 256, 256, 0, stream>>>(g_row, g_col);
    global_edges_kernel<<<(EGLOB + 255) / 256, 256, 0, stream>>>(eidx, grow, gcol);

    const long gN = NGRP, nN = NNODE;
    const long gE = EGRP, nE = EGLOB;

    // 3) group GNN (2 layers) on (512 nodes, 7680 edges)
    zero_kernel<<<(int)((gN * 32 + 255) / 256), 256, 0, stream>>>(sbuf, gN * 32);
    zero_kernel<<<(int)((gN + 255) / 256), 256, 0, stream>>>(cnt, gN);
    edge_scatter_kernel<<<(int)((gE * 32 + 255) / 256), 256, 0, stream>>>(
        g_x, 44, g_ew, EDGE_H, g_row, g_col, F(17), F(18), sbuf, cnt, gE);
    node_mlp_kernel<<<(int)((gN * 32 + 255) / 256), 256, 0, stream>>>(
        g_x, 44, sbuf, cnt, F(19), F(20), gx1, gN);

    zero_kernel<<<(int)((gN * 32 + 255) / 256), 256, 0, stream>>>(sbuf, gN * 32);
    zero_kernel<<<(int)((gN + 255) / 256), 256, 0, stream>>>(cnt, gN);
    edge_scatter_kernel<<<(int)((gE * 32 + 255) / 256), 256, 0, stream>>>(
        gx1, 32, g_ew, EDGE_H, g_row, g_col, F(21), F(22), sbuf, cnt, gE);
    node_mlp_kernel<<<(int)((gN * 32 + 255) / 256), 256, 0, stream>>>(
        gx1, 32, sbuf, cnt, F(23), F(24), gx2, gN);

    // 4) new_x = concat(xh, wsm . gx2)
    combine_kernel<<<(NNODE + 255) / 256, 256, 0, stream>>>(xh, wsm, gx2, bufA);

    // 5) global GNN (2 layers) on (32768 nodes, 524288 edges), edge_attr = edge_w (dim 1)
    zero_kernel<<<(int)((nN * 32 + 255) / 256), 256, 0, stream>>>(sbuf, nN * 32);
    zero_kernel<<<(int)((nN + 255) / 256), 256, 0, stream>>>(cnt, nN);
    edge_scatter_kernel<<<(int)((nE * 32 + 255) / 256), 256, 0, stream>>>(
        bufA, 64, edge_w, 1, grow, gcol, F(25), F(26), sbuf, cnt, nE);
    node_mlp_kernel<<<(int)((nN * 32 + 255) / 256), 256, 0, stream>>>(
        bufA, 64, sbuf, cnt, F(27), F(28), bufB, nN);

    zero_kernel<<<(int)((nN * 32 + 255) / 256), 256, 0, stream>>>(sbuf, nN * 32);
    zero_kernel<<<(int)((nN + 255) / 256), 256, 0, stream>>>(cnt, nN);
    edge_scatter_kernel<<<(int)((nE * 32 + 255) / 256), 256, 0, stream>>>(
        bufB, 32, edge_w, 1, grow, gcol, F(29), F(30), sbuf, cnt, nE);
    node_mlp_kernel<<<(int)((nN * 32 + 255) / 256), 256, 0, stream>>>(
        bufB, 32, sbuf, cnt, F(31), F(32), bufC, nN);

    // 6) decoder: dx = lin(nx), dg = raw_w . dx
    linear_kernel<<<(int)((nN * 32 + 255) / 256), 256, 0, stream>>>(
        bufC, F(33), F(34), dx, nN, 32, 32, 0);
    einsum_cg_kernel<<<Bb * Gg, 64, 0, stream>>>(wraw, dx, dg, 32);

    // 7) decoder group GNN (2 layers), edge_attr = g_ew
    zero_kernel<<<(int)((gN * 32 + 255) / 256), 256, 0, stream>>>(sbuf, gN * 32);
    zero_kernel<<<(int)((gN + 255) / 256), 256, 0, stream>>>(cnt, gN);
    edge_scatter_kernel<<<(int)((gE * 32 + 255) / 256), 256, 0, stream>>>(
        dg, 32, g_ew, EDGE_H, g_row, g_col, F(35), F(36), sbuf, cnt, gE);
    node_mlp_kernel<<<(int)((gN * 32 + 255) / 256), 256, 0, stream>>>(
        dg, 32, sbuf, cnt, F(37), F(38), gx1, gN);

    zero_kernel<<<(int)((gN * 32 + 255) / 256), 256, 0, stream>>>(sbuf, gN * 32);
    zero_kernel<<<(int)((gN + 255) / 256), 256, 0, stream>>>(cnt, gN);
    edge_scatter_kernel<<<(int)((gE * 32 + 255) / 256), 256, 0, stream>>>(
        gx1, 32, g_ew, EDGE_H, g_row, g_col, F(39), F(40), sbuf, cnt, gE);
    node_mlp_kernel<<<(int)((gN * 32 + 255) / 256), 256, 0, stream>>>(
        gx1, 32, sbuf, cnt, F(41), F(42), gx2, gN);

    // 8) dn = concat(dx, raw_w . gx2)
    combine_kernel<<<(NNODE + 255) / 256, 256, 0, stream>>>(dx, wraw, gx2, bufA);

    // 9) decoder global GNN (2 layers)
    zero_kernel<<<(int)((nN * 32 + 255) / 256), 256, 0, stream>>>(sbuf, nN * 32);
    zero_kernel<<<(int)((nN + 255) / 256), 256, 0, stream>>>(cnt, nN);
    edge_scatter_kernel<<<(int)((nE * 32 + 255) / 256), 256, 0, stream>>>(
        bufA, 64, edge_w, 1, grow, gcol, F(43), F(44), sbuf, cnt, nE);
    node_mlp_kernel<<<(int)((nN * 32 + 255) / 256), 256, 0, stream>>>(
        bufA, 64, sbuf, cnt, F(45), F(46), bufB, nN);

    zero_kernel<<<(int)((nN * 32 + 255) / 256), 256, 0, stream>>>(sbuf, nN * 32);
    zero_kernel<<<(int)((nN + 255) / 256), 256, 0, stream>>>(cnt, nN);
    edge_scatter_kernel<<<(int)((nE * 32 + 255) / 256), 256, 0, stream>>>(
        bufB, 32, edge_w, 1, grow, gcol, F(47), F(48), sbuf, cnt, nE);
    node_mlp_kernel<<<(int)((nN * 32 + 255) / 256), 256, 0, stream>>>(
        bufB, 32, sbuf, cnt, F(49), F(50), bufC, nN);

    // 10) prediction head: relu(lin 32->16), relu(lin 16->24)
    linear_kernel<<<(int)((nN * 16 + 255) / 256), 256, 0, stream>>>(
        bufC, F(51), F(52), bufB, nN, 32, 16, 1);
    linear_kernel<<<(int)((nN * 24 + 255) / 256), 256, 0, stream>>>(
        bufB, F(53), F(54), outp, nN, 16, 24, 1);
}
